// CarafeUpsampler_40561671143484
// MI455X (gfx1250) — matrix operand order repaired and checked by reading, not compiled, not run on hardware
//
#include <hip/hip_runtime.h>
#include <hip/hip_bf16.h>
#include <math.h>

typedef _Float16 v16h __attribute__((ext_vector_type(16)));
typedef _Float16 v8h  __attribute__((ext_vector_type(8)));
typedef float    v8f  __attribute__((ext_vector_type(8)));

// Problem constants
static constexpr int NB   = 8;            // batch
static constexpr int C    = 256;          // source channels
static constexpr int H    = 64, W = 64;   // low-res spatial
static constexpr int HW   = H * W;        // 4096
static constexpr int P    = NB * HW;      // 32768 pixels (GEMM M)
static constexpr int CM   = 64;           // compressed channels
static constexpr int ME   = 100;          // encoder out channels (25*4)
static constexpr int MEP  = 112;          // padded to 7 WMMA col tiles
static constexpr int KE   = 576;          // encoder GEMM K (9 taps * 64)
static constexpr int K2   = 25;           // 5x5 kernel taps

// ---------------------------------------------------------------------------
// Pack: source NCHW f32 -> A [pixel][channel] f16 (K-contiguous GEMM-A rows)
// LDS-tiled so both global read (along w) and write (along c) are coalesced.
// ---------------------------------------------------------------------------
__global__ void pack_source(const float* __restrict__ src, _Float16* __restrict__ A) {
  __shared__ _Float16 tile[64][65];
  int nh = blockIdx.x;                 // n*64 + h
  int n = nh >> 6, h = nh & 63;
  int cbase = blockIdx.y * 64;
  int w = threadIdx.x & 63;
  int q = threadIdx.x >> 6;            // 0..3
  for (int cc = q; cc < 64; cc += 4)
    tile[cc][w] = (_Float16)src[((size_t)(n * C + cbase + cc) * H + h) * W + w];
  __syncthreads();
  int c = threadIdx.x & 63;
  for (int ww = q; ww < 64; ww += 4)
    A[((size_t)(nh * 64 + ww)) * C + cbase + c] = tile[c][ww];
}

// Wc [64][256] f32 -> f16 (already [o][k] K-contiguous = B-transpose)
__global__ void pack_wc(const float* __restrict__ Wc, _Float16* __restrict__ Bt) {
  int i = blockIdx.x * 256 + threadIdx.x;
  if (i < CM * C) Bt[i] = (_Float16)Wc[i];
}

// We [100][64][3][3] f32 -> Bt [112][576] f16, k = (ky*3+kx)*64 + c, zero-padded
__global__ void pack_we(const float* __restrict__ We, _Float16* __restrict__ Bt) {
  int i = blockIdx.x * 256 + threadIdx.x;
  if (i >= MEP * KE) return;
  int o = i / KE, k = i - o * KE;
  int tap = k >> 6, c = k & 63;
  int ky = tap / 3, kx = tap - ky * 3;
  _Float16 v = (_Float16)0.f;
  if (o < ME) v = (_Float16)We[((size_t)(o * 64 + c) * 3 + ky) * 3 + kx];
  Bt[(size_t)o * KE + k] = v;
}

// ---------------------------------------------------------------------------
// GEMM1: comp[p][64] = A[p][256] x Wc^T, WMMA f16 16x16x32, f32 accum.
// 4 waves/block; wave owns 16 rows x 64 cols (4 tiles). 512 blocks.
// ---------------------------------------------------------------------------
__global__ void __launch_bounds__(128) gemm_compress(const _Float16* __restrict__ A,
                                                     const _Float16* __restrict__ Bt,
                                                     const float* __restrict__ bias,
                                                     _Float16* __restrict__ comp) {
  int wave = threadIdx.x >> 5;
  int lane = threadIdx.x & 31;
  int l15  = lane & 15;
  int hi   = lane >> 4;                  // lane half selects K sub-pattern
  int mBase = blockIdx.x * 64 + wave * 16;
  const _Float16* arow = A + (size_t)(mBase + l15) * C;

  v8f acc[4] = {};
  for (int kb = 0; kb < C; kb += 32) {
    // A frag (16x32 f16): lanes 0-15 hold K kb..kb+15,
    // lanes 16-31 hold K kb+16..kb+31 (matches B)
    const _Float16* ap = arow + kb + hi * 16;
    __builtin_prefetch(ap + 32, 0, 0);
    v16h af = *(const v16h*)ap;

#pragma unroll
    for (int t = 0; t < 4; ++t) {
      // B frag (32x16 f16): lane = column, lanes 0-15 K 0..15, 16-31 K 16..31
      int col = t * 16 + l15;
      v16h bf = *(const v16h*)(Bt + (size_t)col * C + kb + hi * 16);
      acc[t] = __builtin_amdgcn_wmma_f32_16x16x32_f16(
          false, af, false, bf, (short)0, acc[t], false, false);
    }
  }
#pragma unroll
  for (int t = 0; t < 4; ++t) {
    int col = t * 16 + l15;
    float b = bias[col];
#pragma unroll
    for (int i = 0; i < 8; ++i) {          // VGPR i: M=i (lo lanes) / M=8+i (hi)
      int m = mBase + i + hi * 8;
      comp[(size_t)m * CM + col] = (_Float16)(acc[t][i] + b);
    }
  }
}

// ---------------------------------------------------------------------------
// GEMM2: raw[p][112] = im2col(comp)[p][576] x We^T, same WMMA skeleton.
// Each 32-wide K-chunk sits inside one 3x3 tap (taps are 64 channels).
// ---------------------------------------------------------------------------
__global__ void __launch_bounds__(128) gemm_encode(const _Float16* __restrict__ comp,
                                                   const _Float16* __restrict__ Bt,
                                                   const float* __restrict__ be,
                                                   float* __restrict__ raw) {
  int wave = threadIdx.x >> 5;
  int lane = threadIdx.x & 31;
  int l15  = lane & 15;
  int hi   = lane >> 4;
  int mBase = blockIdx.x * 64 + wave * 16;
  int row = mBase + l15;
  int n = row >> 12, hw = row & 4095;
  int h = hw >> 6, w = hw & 63;

  v8f acc[7] = {};
  for (int kb = 0; kb < KE; kb += 32) {
    int tap = kb >> 6;                    // 0..8
    int ky = tap / 3, kx = tap - ky * 3;
    int hy = h + ky - 1, wx = w + kx - 1;
    v16h af;
#pragma unroll
    for (int i = 0; i < 16; ++i) af[i] = (_Float16)0.f;
    if ((unsigned)hy < 64u && (unsigned)wx < 64u) {
      int cfirst = (kb & 63) + hi * 16;
      const _Float16* ap = comp + ((size_t)((n * 64 + hy) * 64 + wx)) * CM + cfirst;
      af = *(const v16h*)ap;
    }
#pragma unroll
    for (int t = 0; t < 7; ++t) {
      int col = t * 16 + l15;
      v16h bf = *(const v16h*)(Bt + (size_t)col * KE + kb + hi * 16);
      acc[t] = __builtin_amdgcn_wmma_f32_16x16x32_f16(
          false, af, false, bf, (short)0, acc[t], false, false);
    }
  }
#pragma unroll
  for (int t = 0; t < 7; ++t) {
    int col = t * 16 + l15;
    if (col < ME) {
      float b = be[col];
#pragma unroll
      for (int i = 0; i < 8; ++i) {
        int m = mBase + i + hi * 8;
        raw[(size_t)m * MEP + col] = acc[t][i] + b;
      }
    }
  }
}

// ---------------------------------------------------------------------------
// Softmax over 25 taps + pixel-shuffle reindex.
// raw[p][me], me = k2*4 + s  ->  mask[p][s][k2], softmax over k2.
// ---------------------------------------------------------------------------
__global__ void softmax_shuffle(const float* __restrict__ raw, float* __restrict__ mask) {
  int idx = blockIdx.x * blockDim.x + threadIdx.x;
  if (idx >= P * 4) return;
  int p = idx >> 2, s = idx & 3;
  const float* rp = raw + (size_t)p * MEP;
  float vals[K2];
  float mx = -INFINITY;
#pragma unroll
  for (int k = 0; k < K2; ++k) { vals[k] = rp[k * 4 + s]; mx = fmaxf(mx, vals[k]); }
  float sum = 0.f;
#pragma unroll
  for (int k = 0; k < K2; ++k) { vals[k] = __expf(vals[k] - mx); sum += vals[k]; }
  float inv = 1.f / sum;
  float* op = mask + ((size_t)p * 4 + s) * K2;
#pragma unroll
  for (int k = 0; k < K2; ++k) op[k] = vals[k] * inv;
}

// ---------------------------------------------------------------------------
// CARAFE combine: out[n][c][2h+a][2w+b] = sum_k2 src[n][c][h+dy-2][w+dx-2]*mask.
// Block = one (n,h) row of 64 pixels x 4 channels. The row's mask slice
// (64*4*25 floats = 25.6 KB) is DMA'd into LDS via the CDNA5 async
// global->LDS path (ASYNCcnt-tracked, no VGPR round-trip), then reused by
// all 256 threads and all 64 channel-blocks' waves.
// ---------------------------------------------------------------------------
__global__ void __launch_bounds__(256) carafe_combine(const float* __restrict__ src,
                                                      const float* __restrict__ mask,
                                                      float* __restrict__ out) {
  __shared__ float mtile[64 * 100];       // single shared array -> LDS offset 0
  int nh = blockIdx.x;                    // n*64 + h
  int n = nh >> 6, h = nh & 63;
  const float* mrow = mask + (size_t)nh * 64 * 100;

  // 1600 x b128 async copies, EXEC-masked tail.
  for (int i = threadIdx.x; i < 1600; i += 256) {
    unsigned lds_off = (unsigned)(i * 16);
    unsigned long long ga = (unsigned long long)(mrow + i * 4);
    asm volatile("global_load_async_to_lds_b128 %0, %1, off"
                 :: "v"(lds_off), "v"(ga) : "memory");
  }
  asm volatile("s_wait_asynccnt 0x0" ::: "memory");
  __syncthreads();

  int w = threadIdx.x & 63;
  int csub = threadIdx.x >> 6;
  int c = blockIdx.y * 4 + csub;
  const float* sp = src + (size_t)(n * C + c) * HW;
  const float* mp = mtile + w * 100;

  float a0 = 0.f, a1 = 0.f, a2 = 0.f, a3 = 0.f;
#pragma unroll
  for (int dy = 0; dy < 5; ++dy) {
    int hy = h + dy - 2;
    bool oky = (unsigned)hy < 64u;
    const float* srow = sp + hy * 64;
#pragma unroll
    for (int dx = 0; dx < 5; ++dx) {
      int wx = w + dx - 2;
      float v = (oky && (unsigned)wx < 64u) ? srow[wx] : 0.f;
      int k = dy * 5 + dx;
      a0 = fmaf(v, mp[k],      a0);
      a1 = fmaf(v, mp[25 + k], a1);
      a2 = fmaf(v, mp[50 + k], a2);
      a3 = fmaf(v, mp[75 + k], a3);
    }
  }
  float* op = out + (size_t)(n * C + c) * (128 * 128) + (h * 2) * 128 + w * 2;
  op[0]   = a0;
  op[1]   = a1;
  op[128] = a2;
  op[129] = a3;
}

// ---------------------------------------------------------------------------
extern "C" void kernel_launch(void* const* d_in, const int* in_sizes, int n_in,
                              void* d_out, int out_size, void* d_ws, size_t ws_size,
                              hipStream_t stream) {
  const float* source = (const float*)d_in[0];
  // d_in[1] = guidance (unused by forward)
  const float* Wc = (const float*)d_in[2];
  const float* bc = (const float*)d_in[3];
  const float* We = (const float*)d_in[4];
  const float* be = (const float*)d_in[5];
  float* out = (float*)d_out;

  char* ws = (char*)d_ws;
  size_t off = 0;
  auto take = [&](size_t bytes) -> void* {
    void* p = ws + off;
    off += (bytes + 255) & ~(size_t)255;
    return p;
  };
  _Float16* A16    = (_Float16*)take((size_t)P * C * sizeof(_Float16));     // 16.8 MB
  _Float16* Wc16   = (_Float16*)take((size_t)CM * C * sizeof(_Float16));    // 32 KB
  _Float16* comp16 = (_Float16*)take((size_t)P * CM * sizeof(_Float16));    // 4.2 MB
  _Float16* We16   = (_Float16*)take((size_t)MEP * KE * sizeof(_Float16));  // 126 KB
  float*    raw    = (float*)take((size_t)P * MEP * sizeof(float));         // 14.7 MB
  float*    maskF  = (float*)take((size_t)P * 4 * K2 * sizeof(float));      // 13.1 MB
  (void)ws_size; (void)in_sizes; (void)n_in; (void)out_size;

  // 1) packs
  pack_source<<<dim3(NB * H, C / 64), 256, 0, stream>>>(source, A16);
  pack_wc<<<(CM * C + 255) / 256, 256, 0, stream>>>(Wc, Wc16);
  pack_we<<<(MEP * KE + 255) / 256, 256, 0, stream>>>(We, We16);

  // 2) channel compressor GEMM (WMMA f16)
  gemm_compress<<<P / 64, 128, 0, stream>>>(A16, Wc16, bc, comp16);

  // 3) content encoder 3x3 conv as implicit-im2col GEMM (WMMA f16)
  gemm_encode<<<P / 64, 128, 0, stream>>>(comp16, We16, be, raw);

  // 4) softmax over 25 taps + pixel shuffle
  softmax_shuffle<<<(P * 4 + 255) / 256, 256, 0, stream>>>(raw, maskF);

  // 5) CARAFE reassembly (HBM-bound: writes the 134 MB output)
  carafe_combine<<<dim3(NB * H, C / 4), 256, 0, stream>>>(source, maskF, out);
}
